// Decoder_23235773071680
// MI455X (gfx1250) — compile-verified, matrix operand order checked by reading, not run on hardware
//
#include <hip/hip_runtime.h>

// Problem constants (fixed by the reference)
#define BB 8
#define NN 48
#define TT 10
#define DDIM 4
#define HH 256
#define KTY 4
#define EE (NN * (NN - 1))   // 2256 edges
#define H2 (2 * HH)          // 512
#define ME (BB * EE)         // 18048 edge rows (multiple of 16)
#define MN (BB * NN)         // 384 node rows  (multiple of 16)

typedef __bf16 bf16x16 __attribute__((ext_vector_type(16)));
typedef __bf16 bf16x8  __attribute__((ext_vector_type(8)));
typedef float  f32x8   __attribute__((ext_vector_type(8)));

#define F32X8_ZERO {0.f, 0.f, 0.f, 0.f, 0.f, 0.f, 0.f, 0.f}

// Per-lane fragment load for 16-bit WMMA operands (A row-major [M x K],
// B as W^T row-major [N x K]). Lane holds 16 bf16: elements 0..7 at K = k0+8h..,
// elements 8..15 at K = k0+16+8h.. -> two contiguous 16B loads.
static __device__ inline bf16x16 ldfrag(const __bf16* p) {
    union { bf16x8 h[2]; bf16x16 v; } u;
    u.h[0] = *(const bf16x8*)(p);
    u.h[1] = *(const bf16x8*)(p + 16);
    return u.v;
}

static __device__ inline f32x8 wmma_bf16(bf16x16 a, bf16x16 b, f32x8 c) {
    return __builtin_amdgcn_wmma_f32_16x16x32_bf16(
        /*neg_a=*/false, a, /*neg_b=*/false, b,
        /*c_mod=*/(short)0, c, /*reuse_a=*/false, /*reuse_b=*/false);
}

static __device__ inline float sigmoidf_(float x) { return 1.f / (1.f + expf(-x)); }

// ---------------------------------------------------------------- setup kernels

__global__ void k_build_idx(const float* __restrict__ rel_rec,
                            const float* __restrict__ rel_send,
                            int* __restrict__ recv_idx, int* __restrict__ send_idx) {
    int e = blockIdx.x * blockDim.x + threadIdx.x;
    if (e >= EE) return;
    int r = 0, s = 0;
    for (int n = 0; n < NN; ++n) {
        if (rel_rec[e * NN + n]  > 0.5f) r = n;
        if (rel_send[e * NN + n] > 0.5f) s = n;
    }
    recv_idx[e] = r;
    send_idx[e] = s;
}

// dst[c*R + r] = bf16(src[r*C + c])  (weights -> transposed bf16 [out][in])
__global__ void k_transpose_bf16(const float* __restrict__ src, __bf16* __restrict__ dst,
                                 int R, int C) {
    int i = blockIdx.x * blockDim.x + threadIdx.x;
    if (i >= R * C) return;
    int r = i / C, c = i % C;
    dst[c * R + r] = (__bf16)src[i];
}

__global__ void k_zero_f32(float* __restrict__ p, int n) {
    int i = blockIdx.x * blockDim.x + threadIdx.x;
    if (i < n) p[i] = 0.f;
}

// ---------------------------------------------------------------- per-step kernels

// pre[b*E+e][0:H]   = hidden[b][recv(e)][:]
// pre[b*E+e][H:2H]  = hidden[b][send(e)][:]     (f32 -> bf16)
__global__ void k_gather_pre(const float* __restrict__ hidden,
                             const int* __restrict__ recv_idx,
                             const int* __restrict__ send_idx,
                             __bf16* __restrict__ pre) {
    int i = blockIdx.x * blockDim.x + threadIdx.x;
    if (i >= BB * EE * HH) return;
    int h  = i % HH;
    int be = i / HH;
    int e  = be % EE;
    int b  = be / EE;
    pre[(size_t)be * H2 + h]      = (__bf16)hidden[((size_t)b * NN + recv_idx[e]) * HH + h];
    pre[(size_t)be * H2 + HH + h] = (__bf16)hidden[((size_t)b * NN + send_idx[e]) * HH + h];
}

// m1[k] = tanh(pre @ W1[k] + b1[k])  for edge types k=1..3 (stored k-1)
// grid (ME/16, 2), block 256 = 8 waves; wave -> one 16x16 C tile per k.
__global__ void __launch_bounds__(256)
k_edge_fc1(const __bf16* __restrict__ pre, const __bf16* __restrict__ W1t,
           const float* __restrict__ fc1b, __bf16* __restrict__ m1) {
    int wave = threadIdx.x >> 5;
    int lane = threadIdx.x & 31;
    int half = lane >> 4, lr = lane & 15;
    int mt = blockIdx.x;
    int nt = blockIdx.y * 8 + wave;
    int rowA = mt * 16 + lr;
    int rowB = nt * 16 + lr;
    int col  = nt * 16 + lr;
    const __bf16* aRow = pre + (size_t)rowA * H2 + 8 * half;
    for (int k = 0; k < 3; ++k) {
        const __bf16* bRow = W1t + (size_t)k * HH * H2 + (size_t)rowB * H2 + 8 * half;
        f32x8 c = F32X8_ZERO;
        for (int kk = 0; kk < H2; kk += 32) {
            bf16x16 a = ldfrag(aRow + kk);
            bf16x16 b = ldfrag(bRow + kk);
            c = wmma_bf16(a, b, c);
        }
        float bias = fc1b[(k + 1) * HH + col];
        #pragma unroll
        for (int r = 0; r < 8; ++r) {
            int rowC = mt * 16 + r + 8 * half;
            m1[((size_t)k * ME + rowC) * HH + col] = (__bf16)tanhf(c[r] + bias);
        }
    }
}

// msgs[row] = sum_k tanh(m1[k] @ W2[k] + b2[k]) * rel_type[row][k+1] / 3
__global__ void __launch_bounds__(256)
k_edge_fc2(const __bf16* __restrict__ m1, const __bf16* __restrict__ W2t,
           const float* __restrict__ fc2b, const float* __restrict__ rel_type,
           float* __restrict__ msgs) {
    int wave = threadIdx.x >> 5;
    int lane = threadIdx.x & 31;
    int half = lane >> 4, lr = lane & 15;
    int mt = blockIdx.x;
    int nt = blockIdx.y * 8 + wave;
    int rowA = mt * 16 + lr;
    int rowB = nt * 16 + lr;
    int col  = nt * 16 + lr;
    f32x8 cacc[3];
    for (int k = 0; k < 3; ++k) {
        const __bf16* aRow = m1  + ((size_t)k * ME + rowA) * HH + 8 * half;
        const __bf16* bRow = W2t + ((size_t)k * HH + rowB) * HH + 8 * half;
        f32x8 c = F32X8_ZERO;
        for (int kk = 0; kk < HH; kk += 32) {
            bf16x16 a = ldfrag(aRow + kk);
            bf16x16 b = ldfrag(bRow + kk);
            c = wmma_bf16(a, b, c);
        }
        cacc[k] = c;
    }
    float b0 = fc2b[1 * HH + col], b1 = fc2b[2 * HH + col], b2 = fc2b[3 * HH + col];
    #pragma unroll
    for (int r = 0; r < 8; ++r) {
        int row = mt * 16 + r + 8 * half;
        const float* rt = rel_type + (size_t)row * KTY;
        float s = tanhf(cacc[0][r] + b0) * rt[1]
                + tanhf(cacc[1][r] + b1) * rt[2]
                + tanhf(cacc[2][r] + b2) * rt[3];
        msgs[(size_t)row * HH + col] = s * (1.f / 3.f);   // / norm (K-1)
    }
}

// agg[b][n][h] = sum over node n's 47 receiver-contiguous edges of msgs / D.
// (setup enumerates edges receiver-major: edge e = n*(N-1)+c has receiver n)
__global__ void k_aggregate(const float* __restrict__ msgs, __bf16* __restrict__ agg_bf) {
    int i = blockIdx.x * blockDim.x + threadIdx.x;
    if (i >= MN * HH) return;
    int h  = i % HH;
    int bn = i / HH;
    int n  = bn % NN;
    int b  = bn / NN;
    const float* base = msgs + ((size_t)b * EE + (size_t)n * (NN - 1)) * HH + h;
    float s = 0.f;
    for (int c = 0; c < NN - 1; ++c) s += base[(size_t)c * HH];
    agg_bf[(size_t)bn * HH + h] = (__bf16)(s * (1.f / DDIM));
}

// gates[k] = agg @ hid_{r,i,n}_w   (no bias), k = 0,1,2 ; grid (MN/16, 2)
__global__ void __launch_bounds__(256)
k_node_gates(const __bf16* __restrict__ agg, const __bf16* __restrict__ Ht,
             float* __restrict__ gates) {
    int wave = threadIdx.x >> 5;
    int lane = threadIdx.x & 31;
    int half = lane >> 4, lr = lane & 15;
    int mt = blockIdx.x;
    int nt = blockIdx.y * 8 + wave;
    int rowA = mt * 16 + lr;
    int rowB = nt * 16 + lr;
    int col  = nt * 16 + lr;
    const __bf16* aRow = agg + (size_t)rowA * HH + 8 * half;
    for (int k = 0; k < 3; ++k) {
        const __bf16* bRow = Ht + ((size_t)k * HH + rowB) * HH + 8 * half;
        f32x8 c = F32X8_ZERO;
        for (int kk = 0; kk < HH; kk += 32) {
            bf16x16 a = ldfrag(aRow + kk);
            bf16x16 b = ldfrag(bRow + kk);
            c = wmma_bf16(a, b, c);
        }
        #pragma unroll
        for (int r = 0; r < 8; ++r) {
            int rowC = mt * 16 + r + 8 * half;
            gates[((size_t)k * MN + rowC) * HH + col] = c[r];
        }
    }
}

// GRU elementwise update; ins-side D=4 matmuls are scalar (tiny)
__global__ void k_gru(const float* __restrict__ data, const float* __restrict__ gates,
                      const float* __restrict__ in_r_w, const float* __restrict__ in_r_b,
                      const float* __restrict__ in_i_w, const float* __restrict__ in_i_b,
                      const float* __restrict__ in_n_w, const float* __restrict__ in_n_b,
                      float* __restrict__ hidden, __bf16* __restrict__ hid_bf, int t) {
    int i = blockIdx.x * blockDim.x + threadIdx.x;
    if (i >= MN * HH) return;
    int h  = i % HH;
    int bn = i / HH;
    const float* ins = data + ((size_t)bn * TT + t) * DDIM;
    float xr = in_r_b[h], xi = in_i_b[h], xn = in_n_b[h];
    #pragma unroll
    for (int dd = 0; dd < DDIM; ++dd) {
        float x = ins[dd];
        xr += x * in_r_w[dd * HH + h];
        xi += x * in_i_w[dd * HH + h];
        xn += x * in_n_w[dd * HH + h];
    }
    float gr = gates[(size_t)0 * MN * HH + i];
    float gi = gates[(size_t)1 * MN * HH + i];
    float gn = gates[(size_t)2 * MN * HH + i];
    float r  = sigmoidf_(xr + gr);
    float ii = sigmoidf_(xi + gi);
    float ng = tanhf(xn + r * gn);
    float hnew = (1.f - ii) * ng + ii * hidden[i];
    hidden[i] = hnew;
    hid_bf[i] = (__bf16)hnew;
}

// out = relu(A @ W + bias), bf16 out ; M = MN, Kd = Nn = H ; grid (MN/16, 2)
__global__ void __launch_bounds__(256)
k_gemm_relu(const __bf16* __restrict__ A, const __bf16* __restrict__ Wt,
            const float* __restrict__ bias, __bf16* __restrict__ out) {
    int wave = threadIdx.x >> 5;
    int lane = threadIdx.x & 31;
    int half = lane >> 4, lr = lane & 15;
    int mt = blockIdx.x;
    int nt = blockIdx.y * 8 + wave;
    int rowA = mt * 16 + lr;
    int rowB = nt * 16 + lr;
    int col  = nt * 16 + lr;
    const __bf16* aRow = A  + (size_t)rowA * HH + 8 * half;
    const __bf16* bRow = Wt + (size_t)rowB * HH + 8 * half;
    f32x8 c = F32X8_ZERO;
    for (int kk = 0; kk < HH; kk += 32) {
        bf16x16 a = ldfrag(aRow + kk);
        bf16x16 b = ldfrag(bRow + kk);
        c = wmma_bf16(a, b, c);
    }
    float bb = bias[col];
    #pragma unroll
    for (int r = 0; r < 8; ++r) {
        int rowC = mt * 16 + r + 8 * half;
        out[(size_t)rowC * HH + col] = (__bf16)fmaxf(c[r] + bb, 0.f);
    }
}

// pred[b][n][t][d] = ins + p2 @ out3_w + out3_b   (H=256 dot, D=4 -> tiny)
__global__ void k_pred(const float* __restrict__ data, const __bf16* __restrict__ p2,
                       const float* __restrict__ out3_w, const float* __restrict__ out3_b,
                       float* __restrict__ out, int t) {
    int i = blockIdx.x * blockDim.x + threadIdx.x;
    if (i >= MN * DDIM) return;
    int dd = i % DDIM;
    int bn = i / DDIM;
    float acc = out3_b[dd];
    const __bf16* prow = p2 + (size_t)bn * HH;
    for (int h = 0; h < HH; ++h) acc += (float)prow[h] * out3_w[h * DDIM + dd];
    float ins = data[((size_t)bn * TT + t) * DDIM + dd];
    out[((size_t)bn * (TT - 1) + t) * DDIM + dd] = ins + acc;
}

// ---------------------------------------------------------------- host launcher

extern "C" void kernel_launch(void* const* d_in, const int* in_sizes, int n_in,
                              void* d_out, int out_size, void* d_ws, size_t ws_size,
                              hipStream_t stream) {
    (void)in_sizes; (void)n_in; (void)out_size; (void)ws_size;

    const float* data      = (const float*)d_in[0];
    const float* rel_type  = (const float*)d_in[1];
    const float* rel_rec   = (const float*)d_in[2];
    const float* rel_send  = (const float*)d_in[3];
    const float* msg_fc1_w = (const float*)d_in[4];
    const float* msg_fc1_b = (const float*)d_in[5];
    const float* msg_fc2_w = (const float*)d_in[6];
    const float* msg_fc2_b = (const float*)d_in[7];
    const float* hid_r_w   = (const float*)d_in[8];
    const float* hid_i_w   = (const float*)d_in[9];
    const float* hid_n_w   = (const float*)d_in[10];
    const float* in_r_w    = (const float*)d_in[11];
    const float* in_r_b    = (const float*)d_in[12];
    const float* in_i_w    = (const float*)d_in[13];
    const float* in_i_b    = (const float*)d_in[14];
    const float* in_n_w    = (const float*)d_in[15];
    const float* in_n_b    = (const float*)d_in[16];
    const float* out1_w    = (const float*)d_in[17];
    const float* out1_b    = (const float*)d_in[18];
    const float* out2_w    = (const float*)d_in[19];
    const float* out2_b    = (const float*)d_in[20];
    const float* out3_w    = (const float*)d_in[21];
    const float* out3_b    = (const float*)d_in[22];

    // ---- workspace carve-up (deterministic, 256B aligned)
    char* ws = (char*)d_ws;
    size_t off = 0;
    auto walloc = [&](size_t bytes) -> void* {
        void* p = ws + off;
        off += (bytes + 255) & ~(size_t)255;
        return p;
    };
    float*  hidden   = (float*) walloc((size_t)MN * HH * 4);
    __bf16* hid_bf   = (__bf16*)walloc((size_t)MN * HH * 2);
    __bf16* W1t      = (__bf16*)walloc((size_t)3 * HH * H2 * 2);
    __bf16* W2t      = (__bf16*)walloc((size_t)3 * HH * HH * 2);
    __bf16* Ht       = (__bf16*)walloc((size_t)3 * HH * HH * 2);
    __bf16* O1t      = (__bf16*)walloc((size_t)HH * HH * 2);
    __bf16* O2t      = (__bf16*)walloc((size_t)HH * HH * 2);
    int*    recv_idx = (int*)   walloc((size_t)EE * 4);
    int*    send_idx = (int*)   walloc((size_t)EE * 4);
    __bf16* pre      = (__bf16*)walloc((size_t)ME * H2 * 2);
    __bf16* m1       = (__bf16*)walloc((size_t)3 * ME * HH * 2);
    float*  msgs     = (float*) walloc((size_t)ME * HH * 4);
    __bf16* agg_bf   = (__bf16*)walloc((size_t)MN * HH * 2);
    float*  gates    = (float*) walloc((size_t)3 * MN * HH * 4);
    __bf16* p1       = (__bf16*)walloc((size_t)MN * HH * 2);
    __bf16* p2       = (__bf16*)walloc((size_t)MN * HH * 2);

    // ---- one-time setup (re-run every call for determinism)
    k_build_idx<<<(EE + 255) / 256, 256, 0, stream>>>(rel_rec, rel_send, recv_idx, send_idx);

    int ntr1 = (H2 * HH + 255) / 256;   // 512x256 matrices
    int ntr2 = (HH * HH + 255) / 256;   // 256x256 matrices
    for (int k = 1; k <= 3; ++k) {
        k_transpose_bf16<<<ntr1, 256, 0, stream>>>(msg_fc1_w + (size_t)k * H2 * HH,
                                                   W1t + (size_t)(k - 1) * HH * H2, H2, HH);
        k_transpose_bf16<<<ntr2, 256, 0, stream>>>(msg_fc2_w + (size_t)k * HH * HH,
                                                   W2t + (size_t)(k - 1) * HH * HH, HH, HH);
    }
    k_transpose_bf16<<<ntr2, 256, 0, stream>>>(hid_r_w, Ht + (size_t)0 * HH * HH, HH, HH);
    k_transpose_bf16<<<ntr2, 256, 0, stream>>>(hid_i_w, Ht + (size_t)1 * HH * HH, HH, HH);
    k_transpose_bf16<<<ntr2, 256, 0, stream>>>(hid_n_w, Ht + (size_t)2 * HH * HH, HH, HH);
    k_transpose_bf16<<<ntr2, 256, 0, stream>>>(out1_w, O1t, HH, HH);
    k_transpose_bf16<<<ntr2, 256, 0, stream>>>(out2_w, O2t, HH, HH);

    k_zero_f32<<<(MN * HH + 255) / 256, 256, 0, stream>>>(hidden, MN * HH);

    // ---- 9 sequential steps (stream-ordered)
    dim3 edgeGrid(ME / 16, 2);  // 1128 x 2, 8 waves/block -> 16 N-tiles
    dim3 nodeGrid(MN / 16, 2);  // 24 x 2
    for (int t = 0; t < TT - 1; ++t) {
        k_gather_pre<<<(BB * EE * HH + 255) / 256, 256, 0, stream>>>(hidden, recv_idx, send_idx, pre);
        k_edge_fc1<<<edgeGrid, 256, 0, stream>>>(pre, W1t, msg_fc1_b, m1);
        k_edge_fc2<<<edgeGrid, 256, 0, stream>>>(m1, W2t, msg_fc2_b, rel_type, msgs);
        k_aggregate<<<(MN * HH + 255) / 256, 256, 0, stream>>>(msgs, agg_bf);
        k_node_gates<<<nodeGrid, 256, 0, stream>>>(agg_bf, Ht, gates);
        k_gru<<<(MN * HH + 255) / 256, 256, 0, stream>>>(data, gates,
                                                         in_r_w, in_r_b, in_i_w, in_i_b,
                                                         in_n_w, in_n_b, hidden, hid_bf, t);
        k_gemm_relu<<<nodeGrid, 256, 0, stream>>>(hid_bf, O1t, out1_b, p1);
        k_gemm_relu<<<nodeGrid, 256, 0, stream>>>(p1, O2t, out2_b, p2);
        k_pred<<<(MN * DDIM + 255) / 256, 256, 0, stream>>>(data, p2, out3_w, out3_b,
                                                            (float*)d_out, t);
    }
}
